// CommunityModel_45363444580433
// MI455X (gfx1250) — compile-verified, matrix operand order checked by reading, not run on hardware
//
#include <hip/hip_runtime.h>

// ---- problem constants (match reference) ----
#define N_NODESC 20000
#define FNC 128
#define FEC 64
#define DDC 128
#define TTC 128
#define KCLS 5
#define HHC 4
#define DHC 32
#define NLAYERSC 2
#define FFC 512
#define BBC 32768
#define NTOT (3*BBC)          // 98304 tokens, multiple of 16

typedef __attribute__((ext_vector_type(16))) _Float16 v16h;
typedef __attribute__((ext_vector_type(8)))  _Float16 v8h;
typedef __attribute__((ext_vector_type(8)))  float    v8f;
typedef __attribute__((ext_vector_type(4)))  float    v4f;
typedef __attribute__((ext_vector_type(4)))  int      v4i;

union V16U { v16h v; v8h h[2]; };

// CDNA5 async global->LDS staging path (guarded: falls back to plain copy)
#if defined(__gfx1250__) && defined(__has_builtin)
#if __has_builtin(__builtin_amdgcn_global_load_async_to_lds_b128) && \
    __has_builtin(__builtin_amdgcn_s_wait_asynccnt)
#define HAVE_ASYNC_LDS 1
#endif
#endif

#define GLOBAL_AS __attribute__((address_space(1)))
#define LDS_AS    __attribute__((address_space(3)))

// ------------------------------------------------------------------
// fp32 -> fp16 conversion (weights, and x before the head)
// ------------------------------------------------------------------
__global__ void k_f2h(const float* __restrict__ src, _Float16* __restrict__ dst, int n) {
    int i = blockIdx.x * blockDim.x + threadIdx.x;
    if (i < n) dst[i] = (_Float16)src[i];
}

__global__ void k_zero_i32(int* __restrict__ p, int n) {
    int i = blockIdx.x * blockDim.x + threadIdx.x;
    if (i < n) p[i] = 0;
}

// ------------------------------------------------------------------
// build concatenated index / ts arrays: self=[src,dst,neg] other=[dst,src,src]
// ------------------------------------------------------------------
__global__ void k_build_idx(const int* __restrict__ src, const int* __restrict__ dst,
                            const int* __restrict__ neg, const int* __restrict__ eidx,
                            const float* __restrict__ ts,
                            int* __restrict__ selfI, int* __restrict__ otherI,
                            int* __restrict__ eI, float* __restrict__ tsAll) {
    int i = blockIdx.x * blockDim.x + threadIdx.x;
    if (i >= BBC) return;
    int s = src[i], d = dst[i], g = neg[i], e = eidx[i];
    float t = ts[i];
    selfI[i]        = s; selfI[BBC + i]   = d; selfI[2*BBC + i] = g;
    otherI[i]       = d; otherI[BBC + i]  = s; otherI[2*BBC + i] = s;
    eI[i]           = e; eI[BBC + i]      = e; eI[2*BBC + i]     = e;
    tsAll[i]        = t; tsAll[BBC + i]   = t; tsAll[2*BBC + i]  = t;
}

// segment bucketing (attention is permutation-invariant within a segment,
// so no per-segment ts sort is needed — only membership)
__global__ void k_count(const int* __restrict__ selfI, int* __restrict__ cnt, int n) {
    int i = blockIdx.x * blockDim.x + threadIdx.x;
    if (i < n) atomicAdd(&cnt[selfI[i]], 1);
}
__global__ void k_scan(const int* __restrict__ cnt, int* __restrict__ start) {
    if (blockIdx.x == 0 && threadIdx.x == 0) {
        int acc = 0;
        start[0] = 0;
        for (int i = 0; i < N_NODESC; ++i) { acc += cnt[i]; start[i + 1] = acc; }
    }
}
__global__ void k_scatter(const int* __restrict__ selfI, const int* __restrict__ start,
                          int* __restrict__ cursor, int* __restrict__ tokOfSlot, int n) {
    int i = blockIdx.x * blockDim.x + threadIdx.x;
    if (i < n) {
        int nd = selfI[i];
        int slot = atomicAdd(&cursor[nd], 1);
        tokOfSlot[start[nd] + slot] = i;
    }
}

// ------------------------------------------------------------------
// time encoding -> cols [384,512) of the 512-wide fp16 token-cat buffer
// ------------------------------------------------------------------
__global__ void k_time(const float* __restrict__ tsAll, const float* __restrict__ tw,
                       const float* __restrict__ tb, _Float16* __restrict__ xcat) {
    int i = blockIdx.x * blockDim.x + threadIdx.x;
    if (i >= NTOT * TTC) return;
    int t = i >> 7, j = i & 127;
    float v = cosf(tsAll[t] * tw[j] + tb[j]);
    xcat[(size_t)t * 512 + 384 + j] = (_Float16)v;
}

// ------------------------------------------------------------------
// WMMA GEMM:  out = act( X[Ntok x KD] * W[Nout x KD]^T + bias )  (+ residual)
//   8 waves / 256-thread block; wave -> 16x16 tile; blockIdx.y covers Nout/128.
//   A fragment per §7.12.2: lane group g: halves[0..7]=K(g*8+0..7),
//   halves[8..15]=K(16+g*8+0..7)  -> two contiguous b128 loads from LDS.
//   B fragment symmetric: per-lane fixed output col, contiguous K in a W row.
//   Epilogue: accumulators transposed through LDS -> coalesced b128 stores.
//   MODE: 0 = store fp16 Yh; 1 = store fp32 Yf; 2 = Yf += result (residual)
// ------------------------------------------------------------------
template<int KD, int ACT, int MODE, bool GATHER>
__global__ __launch_bounds__(256)
void k_gemm(const _Float16* __restrict__ X, int ldx,
            const float* __restrict__ Gsrc, const int* __restrict__ gidx, int ldg,
            const _Float16* __restrict__ W,
            const float* __restrict__ bias,
            float* __restrict__ Yf, int ldyf,
            _Float16* __restrict__ Yh, int ldyh, int colh,
            int Ntok) {
    alignas(16) __shared__ _Float16 As[16 * KD];
    alignas(16) __shared__ float    Cs[16 * 128];
    const int tile = blockIdx.x * 16;
    const int nb0  = blockIdx.y * 128;

    // ---- stage 16-token A tile into LDS ----
    if (GATHER) {
        // gather fp32 rows, convert to fp16: 8 cols per thread
        for (int i = threadIdx.x; i < 16 * (KD / 8); i += 256) {
            int r = i / (KD / 8), c = (i - r * (KD / 8)) * 8;
            const float* gp = Gsrc + (size_t)gidx[tile + r] * ldg + c;
            v4f f0 = *(const v4f*)gp;
            v4f f1 = *(const v4f*)(gp + 4);
            v8h hv;
#pragma unroll
            for (int j = 0; j < 4; ++j) { hv[j] = (_Float16)f0[j]; hv[4 + j] = (_Float16)f1[j]; }
            *(v8h*)(As + r * KD + c) = hv;
        }
    } else {
#if HAVE_ASYNC_LDS
        for (int i = threadIdx.x; i < 16 * (KD / 8); i += 256) {
            int r = i / (KD / 8), c = (i - r * (KD / 8)) * 8;
            __builtin_amdgcn_global_load_async_to_lds_b128(
                (GLOBAL_AS v4i*)(X + (size_t)(tile + r) * ldx + c),
                (LDS_AS v4i*)(As + r * KD + c), 0, 0);
        }
        __builtin_amdgcn_s_wait_asynccnt(0);
#else
        for (int i = threadIdx.x; i < 16 * (KD / 8); i += 256) {
            int r = i / (KD / 8), c = (i - r * (KD / 8)) * 8;
            *(v8h*)(As + r * KD + c) = *(const v8h*)(X + (size_t)(tile + r) * ldx + c);
        }
#endif
    }
    __syncthreads();

    const int wave = threadIdx.x >> 5;
    const int lane = threadIdx.x & 31;
    const int g    = lane >> 4;
    const int l15  = lane & 15;

    const _Float16* __restrict__ wrow = W + (size_t)(nb0 + wave * 16 + l15) * KD;
    const _Float16* __restrict__ arow = As + l15 * KD;

    v8f acc = {};
#pragma unroll
    for (int kb = 0; kb < KD; kb += 32) {
        V16U a, b;
        a.h[0] = *(const v8h*)(arow + kb + g * 8);
        a.h[1] = *(const v8h*)(arow + kb + 16 + g * 8);
        b.h[0] = *(const v8h*)(wrow + kb + g * 8);
        b.h[1] = *(const v8h*)(wrow + kb + 16 + g * 8);
        acc = __builtin_amdgcn_wmma_f32_16x16x32_f16(false, a.v, false, b.v,
                                                     (short)0, acc, false, false);
    }

    // ---- bias + activation, accumulators -> LDS tile (16 rows x 128 cols) ----
    {
        const int   cloc = wave * 16 + l15;            // local col in block tile
        const float bn   = bias[nb0 + cloc];
#pragma unroll
        for (int v = 0; v < 8; ++v) {
            int m = v + g * 8;                          // D layout row
            float val = acc[v] + bn;
            if (ACT == 1) val = fmaxf(val, 0.0f);
            if (ACT == 2) val = 0.5f * val * (1.0f + erff(val * 0.70710678118654752f));
            Cs[m * 128 + cloc] = val;
        }
    }
    __syncthreads();

    // ---- coalesced writeout: thread -> 8 consecutive cols of one row ----
    {
        const int r  = threadIdx.x >> 4;               // 0..15
        const int c0 = (threadIdx.x & 15) * 8;         // 0..120
        const int t  = tile + r;
        const float* crow = Cs + r * 128 + c0;
        if (MODE == 0) {
            v8h hv;
#pragma unroll
            for (int j = 0; j < 8; ++j) hv[j] = (_Float16)crow[j];
            *(v8h*)(Yh + (size_t)t * ldyh + colh + nb0 + c0) = hv;
        } else if (MODE == 1) {
            float* yp = Yf + (size_t)t * ldyf + nb0 + c0;
            *(v4f*)yp       = *(const v4f*)crow;
            *(v4f*)(yp + 4) = *(const v4f*)(crow + 4);
        } else {
            float* yp = Yf + (size_t)t * ldyf + nb0 + c0;
            v4f r0 = *(const v4f*)yp       + *(const v4f*)crow;
            v4f r1 = *(const v4f*)(yp + 4) + *(const v4f*)(crow + 4);
            *(v4f*)yp       = r0;
            *(v4f*)(yp + 4) = r1;
        }
    }
}

// ------------------------------------------------------------------
// LayerNorm: one wave per token (128 dims -> 4 per lane)
// ------------------------------------------------------------------
__global__ __launch_bounds__(256)
void k_ln(const float* __restrict__ x, const float* __restrict__ s,
          const float* __restrict__ b, _Float16* __restrict__ out, int ntok) {
    int w    = (blockIdx.x * blockDim.x + threadIdx.x) >> 5;
    int lane = threadIdx.x & 31;
    if (w >= ntok) return;
    const float* row = x + (size_t)w * DDC;
    float v0 = row[lane], v1 = row[lane + 32], v2 = row[lane + 64], v3 = row[lane + 96];
    float sum = v0 + v1 + v2 + v3;
    float sq  = v0 * v0 + v1 * v1 + v2 * v2 + v3 * v3;
    for (int o = 16; o > 0; o >>= 1) {
        sum += __shfl_xor(sum, o, 32);
        sq  += __shfl_xor(sq,  o, 32);
    }
    float mean = sum * (1.0f / 128.0f);
    float var  = sq * (1.0f / 128.0f) - mean * mean;
    float rstd = rsqrtf(var + 1e-5f);
    _Float16* orow = out + (size_t)w * DDC;
#pragma unroll
    for (int i = 0; i < 4; ++i) {
        int c = lane + i * 32;
        float vv = (i == 0 ? v0 : i == 1 ? v1 : i == 2 ? v2 : v3);
        orow[c] = (_Float16)((vv - mean) * rstd * s[c] + b[c]);
    }
}

// ------------------------------------------------------------------
// segment attention: one wave per (token, head); lane = head-dim (DH=32);
// online softmax over the token's segment (order-invariant).
// ------------------------------------------------------------------
__global__ __launch_bounds__(256)
void k_attn(const _Float16* __restrict__ qkv, const int* __restrict__ selfI,
            const int* __restrict__ nodeStart, const int* __restrict__ tokOfSlot,
            _Float16* __restrict__ o16, int ntok) {
    int gw   = (blockIdx.x * blockDim.x + threadIdx.x) >> 5;
    int lane = threadIdx.x & 31;
    int t = gw >> 2;          // / H
    int h = gw & 3;
    if (t >= ntok) return;
    const float scale = 0.17677669529663687f;   // 1/sqrt(DH)
    float q = (float)qkv[(size_t)t * 384 + h * 32 + lane];
    int nd = selfI[t];
    int s = nodeStart[nd], e = nodeStart[nd + 1];
    float m = -1e30f, denom = 0.0f, acc = 0.0f;
    for (int slot = s; slot < e; ++slot) {
        int j = tokOfSlot[slot];
        float kd = (float)qkv[(size_t)j * 384 + 128 + h * 32 + lane];
        float p = q * kd;
        for (int o = 16; o > 0; o >>= 1) p += __shfl_xor(p, o, 32);
        p *= scale;
        float nm  = fmaxf(m, p);
        float cor = __expf(m - nm);
        float wgt = __expf(p - nm);
        float vd = (float)qkv[(size_t)j * 384 + 256 + h * 32 + lane];
        acc   = acc * cor + wgt * vd;
        denom = denom * cor + wgt;
        m = nm;
    }
    o16[(size_t)t * 128 + h * 32 + lane] = (_Float16)(acc / denom);
}

// ------------------------------------------------------------------
// head: logits = relu_h(128) . p2_w(5x128) + p2_b; softmax(5) -> out
// one wave per token
// ------------------------------------------------------------------
__global__ __launch_bounds__(256)
void k_out(const _Float16* __restrict__ h, const float* __restrict__ p2w,
           const float* __restrict__ p2b, float* __restrict__ out, int ntok) {
    int w    = (blockIdx.x * blockDim.x + threadIdx.x) >> 5;
    int lane = threadIdx.x & 31;
    if (w >= ntok) return;
    float hv[4];
#pragma unroll
    for (int i = 0; i < 4; ++i) hv[i] = (float)h[(size_t)w * 128 + lane + i * 32];
    float lg[5];
#pragma unroll
    for (int k = 0; k < 5; ++k) {
        float p = 0.0f;
#pragma unroll
        for (int i = 0; i < 4; ++i) p += hv[i] * p2w[k * 128 + lane + i * 32];
        for (int o = 16; o > 0; o >>= 1) p += __shfl_xor(p, o, 32);
        lg[k] = p + p2b[k];
    }
    float mx = fmaxf(fmaxf(fmaxf(lg[0], lg[1]), fmaxf(lg[2], lg[3])), lg[4]);
    float e0 = __expf(lg[0] - mx), e1 = __expf(lg[1] - mx), e2 = __expf(lg[2] - mx),
          e3 = __expf(lg[3] - mx), e4 = __expf(lg[4] - mx);
    float ssum = e0 + e1 + e2 + e3 + e4;
    float r = (lane == 0) ? e0 : (lane == 1) ? e1 : (lane == 2) ? e2 : (lane == 3) ? e3 : e4;
    if (lane < 5) out[(size_t)w * 5 + lane] = r / ssum;
}

// ==================================================================
extern "C" void kernel_launch(void* const* d_in, const int* in_sizes, int n_in,
                              void* d_out, int out_size, void* d_ws, size_t ws_size,
                              hipStream_t stream) {
    (void)in_sizes; (void)n_in; (void)out_size; (void)ws_size;
    const int*   src        = (const int*)  d_in[0];
    const int*   dst        = (const int*)  d_in[1];
    const int*   neg        = (const int*)  d_in[2];
    const int*   eidx       = (const int*)  d_in[3];
    const float* ts         = (const float*)d_in[4];
    const float* node_feats = (const float*)d_in[5];
    const float* edge_feats = (const float*)d_in[6];
    const float* node_lin_w = (const float*)d_in[7];
    const float* node_lin_b = (const float*)d_in[8];
    const float* edge_lin_w = (const float*)d_in[9];
    const float* edge_lin_b = (const float*)d_in[10];
    const float* time_w     = (const float*)d_in[11];
    const float* time_b     = (const float*)d_in[12];
    const float* tok_w      = (const float*)d_in[13];
    const float* tok_b      = (const float*)d_in[14];
    const float* qkv_w      = (const float*)d_in[15];
    const float* qkv_b      = (const float*)d_in[16];
    const float* attn_w     = (const float*)d_in[17];
    const float* attn_b     = (const float*)d_in[18];
    const float* ln1_s      = (const float*)d_in[19];
    const float* ln1_b      = (const float*)d_in[20];
    const float* ln2_s      = (const float*)d_in[21];
    const float* ln2_b      = (const float*)d_in[22];
    const float* ff1_w      = (const float*)d_in[23];
    const float* ff1_b      = (const float*)d_in[24];
    const float* ff2_w      = (const float*)d_in[25];
    const float* ff2_b      = (const float*)d_in[26];
    const float* p1_w       = (const float*)d_in[27];
    const float* p1_b       = (const float*)d_in[28];
    const float* p2_w       = (const float*)d_in[29];
    const float* p2_b       = (const float*)d_in[30];
    float* out = (float*)d_out;

    // ---- workspace carve ----
    char* wp = (char*)d_ws;
    auto carve = [&](size_t bytes) -> void* {
        void* r = (void*)wp;
        wp += (bytes + 255) & ~(size_t)255;
        return r;
    };
    // fp16 weight blob (offsets in halves)
    const size_t oNode = 0;
    const size_t oEdge = oNode + 128 * 128;
    const size_t oTok  = oEdge + 128 * 64;
    const size_t oQkv  = oTok  + 128 * 512;
    const size_t oAttn = oQkv  + 2 * 384 * 128;
    const size_t oFf1  = oAttn + 2 * 128 * 128;
    const size_t oFf2  = oFf1  + 2 * 512 * 128;
    const size_t oP1   = oFf2  + 2 * 128 * 512;
    const size_t oEnd  = oP1   + 128 * 128;
    _Float16* wH     = (_Float16*)carve(oEnd * sizeof(_Float16));
    int*   selfI     = (int*)  carve((size_t)NTOT * 4);
    int*   otherI    = (int*)  carve((size_t)NTOT * 4);
    int*   eI        = (int*)  carve((size_t)NTOT * 4);
    float* tsAll     = (float*)carve((size_t)NTOT * 4);
    int*   nodeCnt   = (int*)  carve((size_t)N_NODESC * 4);
    int*   cursor    = (int*)  carve((size_t)N_NODESC * 4);
    int*   nodeStart = (int*)  carve((size_t)(N_NODESC + 1) * 4);
    int*   tokOfSlot = (int*)  carve((size_t)NTOT * 4);
    _Float16* xcat   = (_Float16*)carve((size_t)NTOT * 512 * 2);   // also ff1 buffer
    float*    xRes   = (float*)   carve((size_t)NTOT * 128 * 4);
    _Float16* h16    = (_Float16*)carve((size_t)NTOT * 128 * 2);
    _Float16* qkv16  = (_Float16*)carve((size_t)NTOT * 384 * 2);

    // ---- convert weights to fp16 ----
    auto cvt = [&](const float* s, size_t off, int n) {
        k_f2h<<<(n + 255) / 256, 256, 0, stream>>>(s, wH + off, n);
    };
    cvt(node_lin_w, oNode, 128 * 128);
    cvt(edge_lin_w, oEdge, 128 * 64);
    cvt(tok_w,      oTok,  128 * 512);
    cvt(qkv_w,      oQkv,  2 * 384 * 128);
    cvt(attn_w,     oAttn, 2 * 128 * 128);
    cvt(ff1_w,      oFf1,  2 * 512 * 128);
    cvt(ff2_w,      oFf2,  2 * 128 * 512);
    cvt(p1_w,       oP1,   128 * 128);

    // ---- indices, segments ----
    k_build_idx<<<(BBC + 255) / 256, 256, 0, stream>>>(src, dst, neg, eidx, ts,
                                                       selfI, otherI, eI, tsAll);
    k_zero_i32<<<(N_NODESC + 255) / 256, 256, 0, stream>>>(nodeCnt, N_NODESC);
    k_zero_i32<<<(N_NODESC + 255) / 256, 256, 0, stream>>>(cursor, N_NODESC);
    k_count<<<(NTOT + 255) / 256, 256, 0, stream>>>(selfI, nodeCnt, NTOT);
    k_scan<<<1, 1, 0, stream>>>(nodeCnt, nodeStart);
    k_scatter<<<(NTOT + 255) / 256, 256, 0, stream>>>(selfI, nodeStart, cursor,
                                                      tokOfSlot, NTOT);

    // ---- token embedding: xcat = [self | other | edge | time] (fp16, 512 wide) ----
    const dim3 blk(256);
    const dim3 g1(NTOT / 16, 1), g3(NTOT / 16, 3), g4(NTOT / 16, 4);
    k_time<<<(NTOT * TTC + 255) / 256, 256, 0, stream>>>(tsAll, time_w, time_b, xcat);
    k_gemm<128, 0, 0, true><<<g1, blk, 0, stream>>>(
        nullptr, 0, node_feats, selfI, FNC, wH + oNode, node_lin_b,
        nullptr, 0, xcat, 512, 0, NTOT);
    k_gemm<128, 0, 0, true><<<g1, blk, 0, stream>>>(
        nullptr, 0, node_feats, otherI, FNC, wH + oNode, node_lin_b,
        nullptr, 0, xcat, 512, 128, NTOT);
    k_gemm<64, 0, 0, true><<<g1, blk, 0, stream>>>(
        nullptr, 0, edge_feats, eI, FEC, wH + oEdge, edge_lin_b,
        nullptr, 0, xcat, 512, 256, NTOT);
    // tok = relu(xcat @ tok_w^T + b)  -> residual stream x (fp32)
    k_gemm<512, 1, 1, false><<<g1, blk, 0, stream>>>(
        xcat, 512, nullptr, nullptr, 0, wH + oTok, tok_b,
        xRes, 128, nullptr, 0, 0, NTOT);

    // ---- transformer layers ----
    const int lnBlocks   = (NTOT * 32 + 255) / 256;
    const int attnBlocks = (NTOT * HHC * 32 + 255) / 256;
    for (int l = 0; l < NLAYERSC; ++l) {
        k_ln<<<lnBlocks, 256, 0, stream>>>(xRes, ln1_s + l * 128, ln1_b + l * 128,
                                           h16, NTOT);
        k_gemm<128, 0, 0, false><<<g3, blk, 0, stream>>>(
            h16, 128, nullptr, nullptr, 0, wH + oQkv + (size_t)l * 384 * 128,
            qkv_b + l * 384, nullptr, 0, qkv16, 384, 0, NTOT);
        k_attn<<<attnBlocks, 256, 0, stream>>>(qkv16, selfI, nodeStart, tokOfSlot,
                                               h16, NTOT);
        k_gemm<128, 0, 2, false><<<g1, blk, 0, stream>>>(
            h16, 128, nullptr, nullptr, 0, wH + oAttn + (size_t)l * 128 * 128,
            attn_b + l * 128, xRes, 128, nullptr, 0, 0, NTOT);
        k_ln<<<lnBlocks, 256, 0, stream>>>(xRes, ln2_s + l * 128, ln2_b + l * 128,
                                           h16, NTOT);
        k_gemm<128, 2, 0, false><<<g4, blk, 0, stream>>>(
            h16, 128, nullptr, nullptr, 0, wH + oFf1 + (size_t)l * 512 * 128,
            ff1_b + l * 512, nullptr, 0, xcat, 512, 0, NTOT);
        k_gemm<512, 0, 2, false><<<g1, blk, 0, stream>>>(
            xcat, 512, nullptr, nullptr, 0, wH + oFf2 + (size_t)l * 128 * 512,
            ff2_b + l * 128, xRes, 128, nullptr, 0, 0, NTOT);
    }

    // ---- head: p1 (relu, WMMA) then 5-way softmax ----
    k_f2h<<<(NTOT * 128 + 255) / 256, 256, 0, stream>>>(xRes, h16, NTOT * 128);
    k_gemm<128, 1, 0, false><<<g1, blk, 0, stream>>>(
        h16, 128, nullptr, nullptr, 0, wH + oP1, p1_b,
        nullptr, 0, qkv16, 128, 0, NTOT);     // reuse qkv16 as 128-wide fp16 buffer
    k_out<<<lnBlocks, 256, 0, stream>>>(qkv16, p2_w, p2_b, out, NTOT);
}
